// KoLeoLoss_12395275616878
// MI455X (gfx1250) — matrix operand order repaired and checked by reading, not compiled, not run on hardware
//
#include <hip/hip_runtime.h>
#include <hip/hip_bf16.h>

typedef __attribute__((ext_vector_type(16))) _Float16 v16h;
typedef __attribute__((ext_vector_type(8)))  _Float16 h8;
typedef __attribute__((ext_vector_type(8)))  float    v8f;

#define N_ROWS 16384
#define DIM 768
#define KT 24                 // 768 / 32 k-steps per 16x16 tile
#define COL_CHUNKS 4
#define TILES_PER_CHUNK 256   // (16384/COL_CHUNKS)/16
#define KOLEO_EPS 1e-8f

// ---------------- Kernel 1: L2-normalize rows, store f16 ----------------
__global__ void __launch_bounds__(256) knorm(const float* __restrict__ x,
                                             _Float16* __restrict__ xnh) {
  const int wave = threadIdx.x >> 5;
  const int lane = threadIdx.x & 31;
  const int row  = blockIdx.x * 8 + wave;
  const float* xr = x + (size_t)row * DIM;
  float vals[24];
  float ss = 0.f;
#pragma unroll
  for (int t = 0; t < 24; ++t) { float v = xr[lane + 32 * t]; vals[t] = v; ss += v * v; }
#pragma unroll
  for (int s = 16; s > 0; s >>= 1) ss += __shfl_xor(ss, s, 32);
  const float inv = 1.0f / fmaxf(sqrtf(ss), KOLEO_EPS);
  _Float16* o = xnh + (size_t)row * DIM;
#pragma unroll
  for (int t = 0; t < 24; ++t) o[lane + 32 * t] = (_Float16)(vals[t] * inv);
}

// ------- Kernel 2: WMMA xn @ xn^T with streaming per-row argmax ---------
// B tiles staged via CDNA5 async global->LDS copies (ASYNCcnt pipeline).
__global__ void __launch_bounds__(256, 1) kargmax(const _Float16* __restrict__ xnh,
                                                  float* __restrict__ bestv,
                                                  int* __restrict__ besti) {
  __shared__ _Float16 Bs[2][16 * DIM];   // 2 x 24 KB double buffer
  const int tid  = threadIdx.x;
  const int wave = tid >> 5;
  const int lane = tid & 31;
  const int rowBase = blockIdx.x * 128 + wave * 16;
  const int ct0 = blockIdx.y * TILES_PER_CHUNK;

  // ---- A fragments: 16 rows x 768 K, resident in VGPRs (ISA A layout) ----
  const int am  = lane & 15;
  const int kba = (lane < 16) ? 0 : 8;
  const _Float16* arow = xnh + (size_t)(rowBase + am) * DIM + kba;
  v16h a[KT];
#pragma unroll
  for (int kk = 0; kk < KT; ++kk) {
    h8 lo = *(const h8*)(arow + kk * 32);
    h8 hi = *(const h8*)(arow + kk * 32 + 16);
    a[kk] = __builtin_shufflevector(lo, hi, 0,1,2,3,4,5,6,7,8,9,10,11,12,13,14,15);
  }

  const int bn  = lane & 15;
  const int kbb = (lane < 16) ? 0 : 8;
  const int gRowBase = rowBase + ((lane >= 16) ? 8 : 0);

  // Per-thread chunk coordinates for tile staging (1536 x 16B chunks / 256 thr)
  const int n0 = (tid + 0 * 256) / 96, k0c = ((tid + 0 * 256) % 96) * 8;
  const int n1 = (tid + 1 * 256) / 96, k1c = ((tid + 1 * 256) % 96) * 8;
  const int n2 = (tid + 2 * 256) / 96, k2c = ((tid + 2 * 256) % 96) * 8;
  const int n3 = (tid + 3 * 256) / 96, k3c = ((tid + 3 * 256) % 96) * 8;
  const int n4 = (tid + 4 * 256) / 96, k4c = ((tid + 4 * 256) % 96) * 8;
  const int n5 = (tid + 5 * 256) / 96, k5c = ((tid + 5 * 256) % 96) * 8;
  const int nArr[6] = {n0, n1, n2, n3, n4, n5};
  const int kArr[6] = {k0c, k1c, k2c, k3c, k4c, k5c};

  float rm[8]; int ri[8];
#pragma unroll
  for (int r = 0; r < 8; ++r) { rm[r] = -3.0f; ri[r] = 0; }

  // Issue async global->LDS copies for one 16x768 f16 tile into buffer `bsel`.
#define ISSUE_TILE(ct_, bsel_)                                                   \
  {                                                                              \
    _Pragma("unroll")                                                            \
    for (int i = 0; i < 6; ++i) {                                                \
      const int n = nArr[i], k = kArr[i];                                        \
      const _Float16* gp = xnh + (size_t)((ct_) * 16 + n) * DIM + k;             \
      const unsigned laddr = (unsigned)(size_t)(&Bs[(bsel_)][n * DIM + k]);      \
      asm volatile("global_load_async_to_lds_b128 %0, %1, off"                   \
                   :: "v"(laddr), "v"(gp) : "memory");                           \
    }                                                                            \
  }

  // ---- prologue: stage first tile into buffer 0 ----
  ISSUE_TILE(ct0, 0);
  asm volatile("s_wait_asynccnt 0x0" ::: "memory");
  __syncthreads();

  int buf = 0;
  for (int t = 0; t < TILES_PER_CHUNK; ++t) {
    const int ct = ct0 + t;
    // Kick async staging of the next tile; overlaps with the WMMA stream.
    if (t + 1 < TILES_PER_CHUNK) ISSUE_TILE(ct + 1, buf ^ 1);

    // ---- 16x16 output tile: 24 x v_wmma_f32_16x16x32_f16 ----
    const _Float16* bp = &Bs[buf][bn * DIM + kbb];
    v8f c = {};
#pragma unroll
    for (int kk = 0; kk < KT; ++kk) {
      h8 blo = *(const h8*)(bp + kk * 32);
      h8 bhi = *(const h8*)(bp + kk * 32 + 16);
      v16h b = __builtin_shufflevector(blo, bhi, 0,1,2,3,4,5,6,7,8,9,10,11,12,13,14,15);
      c = __builtin_amdgcn_wmma_f32_16x16x32_f16(false, a[kk], false, b,
                                                 (short)0, c, false, false);
    }
    // streaming per-row argmax (diagonal excluded)
    const int gCol = ct * 16 + bn;
#pragma unroll
    for (int r = 0; r < 8; ++r) {
      const float v = c[r];
      if ((gRowBase + r != gCol) && v > rm[r]) { rm[r] = v; ri[r] = gCol; }
    }
    // Drain this wave's async copies, then join the workgroup. After this
    // barrier every wave has finished reading Bs[buf] and Bs[buf^1] is fully
    // staged, so one barrier per iteration suffices for the double buffer.
    asm volatile("s_wait_asynccnt 0x0" ::: "memory");
    __syncthreads();
    buf ^= 1;
  }
#undef ISSUE_TILE

  // ---- cross-lane (16-lane group) argmax reduction, tie -> lowest idx ----
#pragma unroll
  for (int r = 0; r < 8; ++r) {
    float v = rm[r]; int idx = ri[r];
#pragma unroll
    for (int s = 1; s < 16; s <<= 1) {
      const float ov = __shfl_xor(v, s, 32);
      const int   oi = __shfl_xor(idx, s, 32);
      if (ov > v || (ov == v && oi < idx)) { v = ov; idx = oi; }
    }
    if ((lane & 15) == 0) {
      const int row = gRowBase + r;
      bestv[blockIdx.y * N_ROWS + row] = v;
      besti[blockIdx.y * N_ROWS + row] = idx;
    }
  }
}

// ------- Kernel 3: merge chunks, exact reference distance per row -------
__global__ void __launch_bounds__(256) kdist(const float* __restrict__ x,
                                             const float* __restrict__ bestv,
                                             const int* __restrict__ besti,
                                             float* __restrict__ partial) {
  const int wave = threadIdx.x >> 5;
  const int lane = threadIdx.x & 31;
  const int row  = blockIdx.x * 8 + wave;

  float bv = bestv[row]; int bi = besti[row];
#pragma unroll
  for (int ch = 1; ch < COL_CHUNKS; ++ch) {
    const float v = bestv[ch * N_ROWS + row];
    const int  ix = besti[ch * N_ROWS + row];
    if (v > bv) { bv = v; bi = ix; }
  }

  const float* xi = x + (size_t)row * DIM;
  const float* xj = x + (size_t)bi * DIM;
  float vi[24], vj[24];
  float ssi = 0.f, ssj = 0.f;
#pragma unroll
  for (int t = 0; t < 24; ++t) {
    const float a = xi[lane + 32 * t]; vi[t] = a; ssi += a * a;
    const float b = xj[lane + 32 * t]; vj[t] = b; ssj += b * b;
  }
#pragma unroll
  for (int s = 16; s > 0; s >>= 1) {
    ssi += __shfl_xor(ssi, s, 32);
    ssj += __shfl_xor(ssj, s, 32);
  }
  const float invi = 1.0f / fmaxf(sqrtf(ssi), KOLEO_EPS);
  const float invj = 1.0f / fmaxf(sqrtf(ssj), KOLEO_EPS);
  float acc = 0.f;
#pragma unroll
  for (int t = 0; t < 24; ++t) {
    const float d = vi[t] * invi - vj[t] * invj + KOLEO_EPS;
    acc += d * d;
  }
#pragma unroll
  for (int s = 16; s > 0; s >>= 1) acc += __shfl_xor(acc, s, 32);
  if (lane == 0) {
    const float dist = sqrtf(acc);
    partial[row] = -logf(dist + KOLEO_EPS) / (float)N_ROWS;
  }
}

// ------- Kernel 4: deterministic fixed-order final reduction -------
__global__ void __launch_bounds__(256) kreduce(const float* __restrict__ partial,
                                               float* __restrict__ out) {
  __shared__ float sm[256];
  const int tid = threadIdx.x;
  float s = 0.f;
  for (int i = 0; i < N_ROWS / 256; ++i) s += partial[tid + i * 256];
  sm[tid] = s;
  __syncthreads();
  for (int w = 128; w > 0; w >>= 1) {
    if (tid < w) sm[tid] += sm[tid + w];
    __syncthreads();
  }
  if (tid == 0) out[0] = sm[0];
}

extern "C" void kernel_launch(void* const* d_in, const int* in_sizes, int n_in,
                              void* d_out, int out_size, void* d_ws, size_t ws_size,
                              hipStream_t stream) {
  const float* x = (const float*)d_in[0];
  char* ws = (char*)d_ws;
  _Float16* xnh   = (_Float16*)ws;                                  // 25,165,824 B
  float*    bestv = (float*)(ws + 25165824);                        //    262,144 B
  int*      besti = (int*)  (ws + 25165824 + 262144);               //    262,144 B
  float*    part  = (float*)(ws + 25165824 + 524288);               //     65,536 B
  float*    out   = (float*)d_out;

  knorm  <<<dim3(N_ROWS / 8),                dim3(256), 0, stream>>>(x, xnh);
  kargmax<<<dim3(N_ROWS / 128, COL_CHUNKS),  dim3(256), 0, stream>>>(xnh, bestv, besti);
  kdist  <<<dim3(N_ROWS / 8),                dim3(256), 0, stream>>>(x, bestv, besti, part);
  kreduce<<<dim3(1),                         dim3(256), 0, stream>>>(part, out);
}
